// MachineFeatureBlock_57939108823289
// MI455X (gfx1250) — matrix operand order repaired and checked by reading, not compiled, not run on hardware
//
#include <hip/hip_runtime.h>

// ---------------------------------------------------------------------------
// MachineFeatureBlock on gfx1250 (CDNA5, wave32, WMMA bf16, async LDS copies)
// B=8, N=2048, DOP=DM=H=256
// ---------------------------------------------------------------------------

typedef __attribute__((ext_vector_type(16))) __bf16 v16bf;
typedef __attribute__((ext_vector_type(8)))  float  v8f;
typedef __attribute__((ext_vector_type(4)))  float  f4v;

#define DEV static __device__ __forceinline__

constexpr int Bsz = 8;
constexpr int Nn  = 2048;
constexpr int Hh  = 256;
constexpr int Kk  = 256;
constexpr int TOT = Bsz * Nn;      // 16384 rows

DEV v8f wmma_bf16(v16bf a, v16bf b, v8f c) {
    // D = A(16x32 bf16) x B(32x16 bf16) + C(16x16 f32)
    return __builtin_amdgcn_wmma_f32_16x16x32_bf16(
        /*neg_a=*/false, a, /*neg_b=*/false, b,
        /*c_mod=*/(short)0, c, /*reuse_a=*/false, /*reuse_b=*/false);
}

// A operand (16x32, 16-bit): lane L holds row L%16; K = {0..7,16..23} for lanes
// 0-15, {8..15,24..31} for lanes 16-31. Source rows are f32, convert to bf16.
DEV v16bf ld_a_f32(const float* p /*row start + k0*/, int hi) {
    const int b0 = hi ? 8 : 0;
    const float4* q0 = (const float4*)(p + b0);
    const float4* q1 = (const float4*)(p + b0 + 16);
    float4 x0 = q0[0], x1 = q0[1], y0 = q1[0], y1 = q1[1];
    float tf[16] = {x0.x,x0.y,x0.z,x0.w, x1.x,x1.y,x1.z,x1.w,
                    y0.x,y0.y,y0.z,y0.w, y1.x,y1.y,y1.z,y1.w};
    union { __bf16 h[16]; v16bf v; } t;
#pragma unroll
    for (int i = 0; i < 16; ++i) t.h[i] = (__bf16)tf[i];
    return t.v;
}

// Same A layout, source already bf16 (row-major rows).
DEV v16bf ld_a_bf16(const __bf16* p /*row start + k0*/, int hi) {
    const int b0 = hi ? 8 : 0;
    union { uint4 u[2]; v16bf v; } t;
    t.u[0] = *(const uint4*)(p + b0);
    t.u[1] = *(const uint4*)(p + b0 + 16);
    return t.v;
}

// B operand (32x16, 16-bit): lane L holds column N=L%16; K = {0..7,16..23} for lanes
// 0-15, {8..15,24..31} for lanes 16-31 (2 bf16 per VGPR), matching the A layout.
DEV v16bf ld_b_bf16(const __bf16* p /*N-row start (K contiguous)*/, int hi) {
    union { uint4 u[2]; v16bf v; } t;
    const int b0 = hi ? 8 : 0;
    t.u[0] = *(const uint4*)(p + b0);
    t.u[1] = *(const uint4*)(p + b0 + 16);
    return t.v;
}

// CDNA5 async global->LDS copy (16B per lane), tracked by ASYNCcnt.
// lds_addr: raw LDS byte address (= low 32 bits of the generic address).
DEV void async_ld_b128(unsigned lds_addr, const __bf16* gaddr) {
    asm volatile("global_load_async_to_lds_b128 %0, %1, off"
                 :: "v"(lds_addr), "v"(gaddr)
                 : "memory");
}
DEV unsigned lds_addr_of(const void* p) {
    return (unsigned)(unsigned long long)(uintptr_t)p;
}

DEV float dot4(float4 a, float4 b) { return a.x*b.x + a.y*b.y + a.z*b.z + a.w*b.w; }

// ---------------------------------------------------------------------------
// 1) Fold score projections: w_sq = q_w.T @ s[:H] etc. (4 vectors + 4 consts)
// ---------------------------------------------------------------------------
__global__ __launch_bounds__(256) void vec_fold_kernel(
    const float* __restrict__ q_w, const float* __restrict__ q_b,
    const float* __restrict__ kv_w, const float* __restrict__ kv_b,
    const float* __restrict__ oma_s,
    const float* __restrict__ sa_w, const float* __restrict__ sa_b,
    const float* __restrict__ sa_s,
    float* __restrict__ vecs)
{
    int d = threadIdx.x;
    float v0 = 0.f, v1 = 0.f, v2 = 0.f, v3 = 0.f;
    for (int h = 0; h < 256; ++h) {
        v0 += q_w[h * 256 + d]          * oma_s[h];
        v1 += kv_w[h * 256 + d]         * oma_s[256 + h];
        v2 += sa_w[h * 256 + d]         * sa_s[h];
        v3 += sa_w[(256 + h) * 256 + d] * sa_s[256 + h];
    }
    vecs[d] = v0; vecs[256 + d] = v1; vecs[512 + d] = v2; vecs[768 + d] = v3;
    if (d < 4) {
        float c = 0.f;
        for (int h = 0; h < 256; ++h) {
            if      (d == 0) c += q_b[h]        * oma_s[h];
            else if (d == 1) c += kv_b[h]       * oma_s[256 + h];
            else if (d == 2) c += sa_b[h]       * sa_s[h];
            else             c += sa_b[256 + h] * sa_s[256 + h];
        }
        vecs[1024 + d] = c;
    }
}

// ---------------------------------------------------------------------------
// 2) Fuse output projection + mix: W1 = mix_w[:,:DM] @ oma_o_w (bf16),
//    W2 = mix_w[:,DM:] @ sa_o_w, bias_out = mix_b + o_b contributions.
// ---------------------------------------------------------------------------
__global__ __launch_bounds__(256) void fuse_weights_kernel(
    const float* __restrict__ mix_w, const float* __restrict__ mix_b,
    const float* __restrict__ o1w, const float* __restrict__ o1b,
    const float* __restrict__ o2w, const float* __restrict__ o2b,
    __bf16* __restrict__ W1, __bf16* __restrict__ W2,
    float* __restrict__ bout)
{
    int j = blockIdx.x, h = threadIdx.x;
    const float* mrow = mix_w + (size_t)j * 512;
    float s1 = 0.f, s2 = 0.f;
    for (int d = 0; d < 256; ++d) {
        s1 += mrow[d]       * o1w[d * 256 + h];
        s2 += mrow[256 + d] * o2w[d * 256 + h];
    }
    W1[j * 256 + h] = (__bf16)s1;
    W2[j * 256 + h] = (__bf16)s2;
    if (h == 0) {
        float c = mix_b[j];
        for (int d = 0; d < 256; ++d)
            c += mrow[d] * o1b[d] + mrow[256 + d] * o2b[d];
        bout[j] = c;
    }
}

// ---------------------------------------------------------------------------
// 3) GEMV scores: sq/sk for both attentions (consts folded in).
// ---------------------------------------------------------------------------
__global__ __launch_bounds__(256) void score_kernel(
    const float* __restrict__ mf, const float* __restrict__ of,
    const float* __restrict__ vecs,
    float* __restrict__ sqo, float* __restrict__ sko,
    float* __restrict__ sqs, float* __restrict__ sks)
{
    int t = blockIdx.x * blockDim.x + threadIdx.x;   // 0 .. TOT-1
    const float* mrow = mf + (size_t)t * 256;
    const float* orow = of + (size_t)t * 256;
    float a0 = 0.f, a1 = 0.f, a2 = 0.f, a3 = 0.f;
#pragma unroll 4
    for (int d = 0; d < 256; d += 4) {
        float4 m4 = *(const float4*)(mrow + d);
        float4 o4 = *(const float4*)(orow + d);
        a0 += dot4(m4, *(const float4*)(vecs + d));
        a1 += dot4(o4, *(const float4*)(vecs + 256 + d));
        a2 += dot4(m4, *(const float4*)(vecs + 512 + d));
        a3 += dot4(m4, *(const float4*)(vecs + 768 + d));
    }
    sqo[t] = a0 + vecs[1024];
    sko[t] = a1 + vecs[1025];
    sqs[t] = a2 + vecs[1026];
    sks[t] = a3 + vecs[1027];
}

// ---------------------------------------------------------------------------
// 4) Transposed V projection: vT[b,h,n] = (feat @ W.T + bias).T as bf16.
//    WMMA with A = weight rows (M=h), B = feature rows (N=n) -> output lands
//    transposed & contiguous with zero shuffling. Features staged in LDS bf16.
//    Block: 128 thr (4 waves) = 64 h-rows x 256 n-cols.
// ---------------------------------------------------------------------------
__global__ __launch_bounds__(128) void proj_vT_kernel(
    const float* __restrict__ feat,  // (TOT, 256)
    const float* __restrict__ W,     // (>=64 rows used) x 256, row-major
    const float* __restrict__ bias,
    __bf16* __restrict__ outT)       // (B, H, N)
{
    __shared__ __align__(16) __bf16 ft[256][40];     // 256 n-rows x 32 k (+pad)

    const int htile = blockIdx.x & 3;
    const int ntile = blockIdx.x >> 2;               // 0..63
    const int wave  = threadIdx.x >> 5;
    const int lane  = threadIdx.x & 31;
    const int hi    = lane >> 4;
    const int h0    = htile * 64 + wave * 16;
    const int n0    = ntile * 256;

    v8f acc[16];
#pragma unroll
    for (int i = 0; i < 16; ++i)
#pragma unroll
        for (int g = 0; g < 8; ++g) acc[i][g] = 0.f;

    for (int k0 = 0; k0 < Kk; k0 += 32) {
        __syncthreads();
#pragma unroll
        for (int rr = 0; rr < 2; ++rr) {
            int r = threadIdx.x + rr * 128;
            const float4* src = (const float4*)(feat + (size_t)(n0 + r) * Kk + k0);
            uint4* dst = (uint4*)&ft[r][0];
#pragma unroll
            for (int i = 0; i < 4; ++i) {
                float4 x = src[2 * i], y = src[2 * i + 1];
                union { __bf16 h[8]; uint4 u; } t;
                t.h[0] = (__bf16)x.x; t.h[1] = (__bf16)x.y;
                t.h[2] = (__bf16)x.z; t.h[3] = (__bf16)x.w;
                t.h[4] = (__bf16)y.x; t.h[5] = (__bf16)y.y;
                t.h[6] = (__bf16)y.z; t.h[7] = (__bf16)y.w;
                dst[i] = t.u;
            }
        }
        __syncthreads();

        v16bf A = ld_a_f32(W + (size_t)(h0 + (lane & 15)) * Kk + k0, hi);
#pragma unroll
        for (int nt = 0; nt < 16; ++nt) {
            v16bf Bv = ld_b_bf16(&ft[nt * 16 + (lane & 15)][0], hi);
            acc[nt] = wmma_bf16(A, Bv, acc[nt]);
        }
    }

    const int b    = ntile >> 3;
    const int nloc = (ntile & 7) * 256;
#pragma unroll
    for (int g = 0; g < 8; ++g) {
        int h = h0 + g + hi * 8;
        float bg = bias[h];
#pragma unroll
        for (int nt = 0; nt < 16; ++nt) {
            int nl = nloc + nt * 16 + (lane & 15);
            outT[((size_t)b * Hh + h) * Nn + nl] = (__bf16)(acc[nt][g] + bg);
        }
    }
}

// ---------------------------------------------------------------------------
// 5) Fused attention: flash-style online softmax over 32-column tiles.
//    score = lrelu(sq[m]+sk[c]+b) * mask[m,c]  (mask streamed once, NT hint)
//    p-tile built in registers in WMMA A-layout; vT tile double-buffered in
//    LDS via CDNA5 global_load_async_to_lds_b128 (ASYNCcnt), overlapping the
//    next tile's copy with this tile's 16 WMMAs.
//    Block: 128 thr (4 waves) = 64 rows; each wave owns 16 rows x H=256.
// ---------------------------------------------------------------------------
DEV void stage_vt_async(const __bf16* vTb, int c0, __bf16 (*buf)[40], int tid) {
#pragma unroll
    for (int rr = 0; rr < 2; ++rr) {
        int h = tid + rr * 128;
        const __bf16* g = vTb + (size_t)h * Nn + c0;
        unsigned l = lds_addr_of(&buf[h][0]);
#pragma unroll
        for (int i = 0; i < 4; ++i)
            async_ld_b128(l + i * 16, g + i * 8);
    }
}

__global__ __launch_bounds__(128) void attn_kernel(
    const float* __restrict__ sq, const float* __restrict__ sk,
    const float* __restrict__ sb_ptr,
    const float* __restrict__ mask,          // (B, N, N)
    const __bf16* __restrict__ vT,           // (B, H, N)
    __bf16* __restrict__ wf)                 // (B, N, H)
{
    __shared__ __align__(16) __bf16 vt_lds[2][Hh][40]; // 2 x 20KB double buffer

    const int b    = blockIdx.x >> 5;
    const int tile = blockIdx.x & 31;
    const int wave = threadIdx.x >> 5;
    const int lane = threadIdx.x & 31;
    const int hi   = lane >> 4;
    const int m0   = tile * 64 + wave * 16;
    const int row  = m0 + (lane & 15);

    const float sb   = sb_ptr[0];
    const float sq_r = sq[b * Nn + row];
    const float* mrow = mask + ((size_t)b * Nn + row) * Nn;
    const float* skb  = sk + b * Nn;
    const __bf16* vTb = vT + (size_t)b * Hh * Nn;

    v8f acc[16];
#pragma unroll
    for (int i = 0; i < 16; ++i)
#pragma unroll
        for (int g = 0; g < 8; ++g) acc[i][g] = 0.f;

    float m_run = -__builtin_inff();
    float l_run = 0.f;
    const int kb = hi ? 8 : 0;
    constexpr int NT = Nn / 32;                       // 64 column tiles

    // prologue: async-copy tile 0 into buffer 0
    stage_vt_async(vTb, 0, vt_lds[0], threadIdx.x);

    for (int it = 0; it < NT; ++it) {
        const int c0 = it * 32;

        // kick off next tile's async copy into the other buffer (safe: the
        // barrier closing iteration it-1 retired all readers of that buffer),
        // then retire this tile's 8 async ops (in-order ASYNCcnt).
        if (it + 1 < NT) {
            stage_vt_async(vTb, c0 + 32, vt_lds[(it + 1) & 1], threadIdx.x);
            asm volatile("s_wait_asynccnt 0x8" ::: "memory");
        } else {
            asm volatile("s_wait_asynccnt 0x0" ::: "memory");
        }
        __syncthreads();
        const __bf16 (*cur)[40] = vt_lds[it & 1];

        if (c0 + 256 < Nn) __builtin_prefetch(mrow + c0 + 256, 0, 1);

        // ---- scores for this lane's 16 (row, c) pairs in A-layout order ----
        const float* mp = mrow + c0 + kb;
        const float* sp = skb + c0 + kb;
        f4v mv0 = __builtin_nontemporal_load((const f4v*)mp);
        f4v mv1 = __builtin_nontemporal_load(((const f4v*)mp) + 1);
        f4v mv2 = __builtin_nontemporal_load((const f4v*)(mp + 16));
        f4v mv3 = __builtin_nontemporal_load(((const f4v*)(mp + 16)) + 1);
        float4 sv0 = ((const float4*)sp)[0];
        float4 sv1 = ((const float4*)sp)[1];
        float4 sv2 = ((const float4*)(sp + 16))[0];
        float4 sv3 = ((const float4*)(sp + 16))[1];
        float mk[16] = {mv0.x,mv0.y,mv0.z,mv0.w, mv1.x,mv1.y,mv1.z,mv1.w,
                        mv2.x,mv2.y,mv2.z,mv2.w, mv3.x,mv3.y,mv3.z,mv3.w};
        float sv[16] = {sv0.x,sv0.y,sv0.z,sv0.w, sv1.x,sv1.y,sv1.z,sv1.w,
                        sv2.x,sv2.y,sv2.z,sv2.w, sv3.x,sv3.y,sv3.z,sv3.w};
        float ps[16];
#pragma unroll
        for (int i = 0; i < 16; ++i) {
            float s = sq_r + sv[i] + sb;
            s = (s > 0.f) ? s : 0.01f * s;    // leaky_relu(0.01)
            ps[i] = s * mk[i];
        }

        // ---- online softmax update ----
        float tmax = ps[0];
#pragma unroll
        for (int i = 1; i < 16; ++i) tmax = fmaxf(tmax, ps[i]);
        tmax = fmaxf(tmax, __shfl_xor(tmax, 16));  // combine lane halves (same row)
        float m_new  = fmaxf(m_run, tmax);
        float factor = __expf(m_run - m_new);
        float psum = 0.f;
#pragma unroll
        for (int i = 0; i < 16; ++i) { ps[i] = __expf(ps[i] - m_new); psum += ps[i]; }
        psum += __shfl_xor(psum, 16);
        l_run = l_run * factor + psum;
        m_run = m_new;

        // ---- rescale accumulators (per C-layout row broadcast) ----
        float facs[8];
#pragma unroll
        for (int g = 0; g < 8; ++g)
            facs[g] = __shfl(factor, g + (hi ? 8 : 0));
#pragma unroll
        for (int nt = 0; nt < 16; ++nt)
#pragma unroll
            for (int g = 0; g < 8; ++g) acc[nt][g] *= facs[g];

        // ---- A = p tile (bf16), B = vT tile from LDS; 16 WMMAs ----
        union { __bf16 h[16]; v16bf v; } Au;
#pragma unroll
        for (int i = 0; i < 16; ++i) Au.h[i] = (__bf16)ps[i];
#pragma unroll
        for (int nt = 0; nt < 16; ++nt) {
            v16bf Bv = ld_b_bf16(&cur[nt * 16 + (lane & 15)][0], hi);
            acc[nt] = wmma_bf16(Au.v, Bv, acc[nt]);
        }
        __syncthreads();   // all waves done reading cur before it is re-staged
    }

    // ---- normalize and store wf (bf16, row-major) ----
    float linv[8];
#pragma unroll
    for (int g = 0; g < 8; ++g) {
        float lg = __shfl(l_run, g + (hi ? 8 : 0));
        linv[g] = 1.f / lg;
    }
#pragma unroll
    for (int g = 0; g < 8; ++g) {
        int r = m0 + g + hi * 8;
#pragma unroll
        for (int nt = 0; nt < 16; ++nt) {
            int hh = nt * 16 + (lane & 15);
            wf[((size_t)b * Nn + r) * Hh + hh] = (__bf16)(acc[nt][g] * linv[g]);
        }
    }
}

// ---------------------------------------------------------------------------
// 6) Final fused GEMM: out = wf_oma @ W1.T + wf_sa @ W2.T + bias  (f32 out)
// ---------------------------------------------------------------------------
__global__ __launch_bounds__(128) void final_gemm_kernel(
    const __bf16* __restrict__ wf1, const __bf16* __restrict__ wf2,
    const __bf16* __restrict__ W1,  const __bf16* __restrict__ W2,
    const float* __restrict__ bias, float* __restrict__ out)
{
    const int wave = threadIdx.x >> 5;
    const int lane = threadIdx.x & 31;
    const int hi   = lane >> 4;
    const int m0   = blockIdx.x * 64 + wave * 16;
    const int row  = m0 + (lane & 15);

    v8f acc[16];
#pragma unroll
    for (int i = 0; i < 16; ++i)
#pragma unroll
        for (int g = 0; g < 8; ++g) acc[i][g] = 0.f;

    for (int phase = 0; phase < 2; ++phase) {
        const __bf16* wf = phase ? wf2 : wf1;
        const __bf16* Wm = phase ? W2  : W1;
        const __bf16* arow = wf + (size_t)row * Hh;
#pragma unroll
        for (int k0 = 0; k0 < Hh; k0 += 32) {
            v16bf A = ld_a_bf16(arow + k0, hi);
#pragma unroll
            for (int nt = 0; nt < 16; ++nt) {
                const __bf16* brow = Wm + (size_t)(nt * 16 + (lane & 15)) * Hh + k0;
                acc[nt] = wmma_bf16(A, ld_b_bf16(brow, hi), acc[nt]);
            }
        }
    }

#pragma unroll
    for (int g = 0; g < 8; ++g) {
        size_t r = (size_t)(m0 + g + hi * 8);
#pragma unroll
        for (int nt = 0; nt < 16; ++nt) {
            int j = nt * 16 + (lane & 15);
            out[r * 256 + j] = acc[nt][g] + bias[j];
        }
    }
}

// ---------------------------------------------------------------------------
// Launcher
// ---------------------------------------------------------------------------
extern "C" void kernel_launch(void* const* d_in, const int* in_sizes, int n_in,
                              void* d_out, int out_size, void* d_ws, size_t ws_size,
                              hipStream_t stream) {
    const float* of     = (const float*)d_in[0];   // (B,N,DOP)
    const float* mf     = (const float*)d_in[1];   // (B,N,DM)
    const float* tmask  = (const float*)d_in[2];   // (B,N,N)
    const float* smask  = (const float*)d_in[3];   // (B,N,N)
    const float* q_w    = (const float*)d_in[4];
    const float* q_b    = (const float*)d_in[5];
    const float* kv_w   = (const float*)d_in[6];
    const float* kv_b   = (const float*)d_in[7];
    const float* oma_s  = (const float*)d_in[8];
    const float* oma_sb = (const float*)d_in[9];
    const float* o1w    = (const float*)d_in[10];
    const float* o1b    = (const float*)d_in[11];
    const float* sa_w   = (const float*)d_in[12];
    const float* sa_b   = (const float*)d_in[13];
    const float* sa_s   = (const float*)d_in[14];
    const float* sa_sb  = (const float*)d_in[15];
    const float* o2w    = (const float*)d_in[16];
    const float* o2b    = (const float*)d_in[17];
    const float* mix_w  = (const float*)d_in[18];
    const float* mix_b  = (const float*)d_in[19];
    float* out = (float*)d_out;

    char* ws = (char*)d_ws;
    size_t off = 0;
    auto alloc = [&](size_t bytes) { char* p = ws + off; off += (bytes + 255) & ~(size_t)255; return p; };

    const size_t mat_bf16 = (size_t)Bsz * Nn * Hh * sizeof(__bf16);   // 8.4 MB
    __bf16* vT_oma = (__bf16*)alloc(mat_bf16);
    __bf16* vT_sa  = (__bf16*)alloc(mat_bf16);
    __bf16* wf_oma = (__bf16*)alloc(mat_bf16);
    __bf16* wf_sa  = (__bf16*)alloc(mat_bf16);
    float* sq_oma  = (float*)alloc((size_t)TOT * 4);
    float* sk_oma  = (float*)alloc((size_t)TOT * 4);
    float* sq_sa   = (float*)alloc((size_t)TOT * 4);
    float* sk_sa   = (float*)alloc((size_t)TOT * 4);
    float* vecs    = (float*)alloc(1028 * 4);
    __bf16* W1     = (__bf16*)alloc(256 * 256 * sizeof(__bf16));
    __bf16* W2     = (__bf16*)alloc(256 * 256 * sizeof(__bf16));
    float* bout    = (float*)alloc(256 * 4);

    // 1) fold score-weight vectors; 2) fuse output-projection weights
    vec_fold_kernel<<<1, 256, 0, stream>>>(q_w, q_b, kv_w, kv_b, oma_s, sa_w, sa_b, sa_s, vecs);
    fuse_weights_kernel<<<256, 256, 0, stream>>>(mix_w, mix_b, o1w, o1b, o2w, o2b, W1, W2, bout);

    // 3) additive-score GEMVs
    score_kernel<<<TOT / 256, 256, 0, stream>>>(mf, of, vecs, sq_oma, sk_oma, sq_sa, sk_sa);

    // 4) V projections (stored transposed bf16): v = of @ kv_w[H:].T, v2 = mf @ sa_w[2H:].T
    proj_vT_kernel<<<256, 128, 0, stream>>>(of, kv_w + (size_t)Hh * Kk,     kv_b + Hh,     vT_oma);
    proj_vT_kernel<<<256, 128, 0, stream>>>(mf, sa_w + (size_t)2 * Hh * Kk, sa_b + 2 * Hh, vT_sa);

    // 5) fused masked-softmax attentions (mask streamed from HBM exactly once)
    attn_kernel<<<256, 128, 0, stream>>>(sq_oma, sk_oma, oma_sb, tmask, vT_oma, wf_oma);
    attn_kernel<<<256, 128, 0, stream>>>(sq_sa,  sk_sa,  sa_sb,  smask, vT_sa,  wf_sa);

    // 6) fused output projection + mix
    final_gemm_kernel<<<256, 128, 0, stream>>>(wf_oma, wf_sa, W1, W2, bout, out);

    (void)in_sizes; (void)n_in; (void)out_size; (void)ws_size;
}
